// TDSPMLPHead_24163486007454
// MI455X (gfx1250) — matrix operand order repaired and checked by reading, not compiled, not run on hardware
//
#include <hip/hip_runtime.h>
#include <math.h>

typedef __attribute__((ext_vector_type(16))) _Float16 v16h;
typedef __attribute__((ext_vector_type(8)))  float    v8f;

#define Bn   8
#define Nn   128
#define Mn   128
#define En   256
#define Hn   512

// ---------------------------------------------------------------------------
// L2-normalize rows of length E=256; emit fp32 and f16 copies.
__global__ __launch_bounds__(256) void prep_norm(const float* __restrict__ X,
                                                 float* __restrict__ Xn,
                                                 _Float16* __restrict__ Xh) {
    __shared__ float red[256];
    const int row = blockIdx.x;
    const int e   = threadIdx.x;
    float v = X[row * En + e];
    red[e] = v * v;
    __syncthreads();
    for (int s = 128; s > 0; s >>= 1) {
        if (e < s) red[e] += red[e + s];
        __syncthreads();
    }
    float nrm = sqrtf(red[0]);
    float inv = 1.0f / fmaxf(nrm, 1e-12f);
    float y = v * inv;
    Xn[row * En + e] = y;
    Xh[row * En + e] = (_Float16)y;
}

// ---------------------------------------------------------------------------
// Partial products: Out[row,h] = sum_e Xn[row,e]*W1[eoff+e,h] (+ b1[h] if addb)
__global__ __launch_bounds__(256) void prep_part(const float* __restrict__ Xn,
                                                 const float* __restrict__ W1,
                                                 const float* __restrict__ b1,
                                                 float* __restrict__ Out,
                                                 int eoff, int addb) {
    __shared__ float sx[256];
    const int row = blockIdx.x >> 1;
    const int h   = ((blockIdx.x & 1) << 8) + threadIdx.x;
    sx[threadIdx.x] = Xn[row * En + threadIdx.x];
    __syncthreads();
    float acc = addb ? b1[h] : 0.0f;
#pragma unroll 4
    for (int e = 0; e < En; ++e)
        acc += sx[e] * W1[(eoff + e) * Hn + h];
    Out[row * Hn + h] = acc;
}

// ---------------------------------------------------------------------------
// Prepack W1 bottom block (rows 512..767) into the 16-bit WMMA B-operand
// lane layout: flat idx = kc*16384 + ct*512 + lane*16 + i, where
// k = kc*32 + ((lane>=16)?8:0) + (i&7) + ((i>=8)?16:0), h = ct*16 + (lane&15).
__global__ __launch_bounds__(256) void prep_bpack(const float* __restrict__ W1,
                                                  _Float16* __restrict__ Bp) {
    const int idx = blockIdx.x * 256 + threadIdx.x;     // 0 .. 131071
    const int i  = idx & 15;
    const int L  = (idx >> 4) & 31;
    const int ct = (idx >> 9) & 31;
    const int kc = idx >> 14;
    const int kl = ((L >> 4) << 3) + (i & 7) + ((i >> 3) << 4); // (L<16?0:8)+(i&7)+(i>=8?16:0)
    const int k  = kc * 32 + kl;
    const int h  = ct * 16 + (L & 15);
    Bp[idx] = (_Float16)W1[(2 * En + k) * Hn + h];
}

// ---------------------------------------------------------------------------
// Main fused kernel: abs-diff GEMM (f16 WMMA, f32 acc) + bias/precomp add +
// LayerNorm + SiLU + W2 dot.  Workgroup = 8 waves: 4 n-values x 2 h-halves,
// 16 m-values.  grid = B * (N/4) * (M/16) = 2048 blocks of 256 threads.
__global__ __launch_bounds__(256) void tdsp_main(
    const _Float16* __restrict__ tnh, const _Float16* __restrict__ dnh,
    const float* __restrict__ Ap,     const float* __restrict__ Dp,
    const _Float16* __restrict__ Bp,
    const float* __restrict__ gamma,  const float* __restrict__ beta,
    const float* __restrict__ W2,     const float* __restrict__ b2,
    float* __restrict__ out) {

    __shared__ _Float16 sT[4 * 256];                 //  2 KB  normalized t rows (f16)
    __shared__ _Float16 sD[16 * 256];                //  8 KB  normalized d rows (f16)
    __shared__ float    sAp[4 * 512];                //  8 KB  t-part + b1
    __shared__ __align__(32) _Float16 sB[32 * 512];  // 32 KB  one K-chunk of packed W1b
    __shared__ float    sG[512], sBt[512], sW2[512]; //  6 KB
    __shared__ float    sRed[8][16][2];              //  1 KB  LN partial sums
    __shared__ float    sOut[8][16];                 // .5 KB  W2-dot partials

    const int tid   = threadIdx.x;
    const int wave  = tid >> 5;
    const int lane  = tid & 31;
    const int nl    = wave & 3;        // which n within tile
    const int hh    = wave >> 2;       // h half (0: h<256, 1: h>=256)
    const int b     = blockIdx.x >> 8;
    const int rem   = blockIdx.x & 255;
    const int n0    = (rem >> 3) * 4;
    const int m0    = (rem & 7) * 16;
    const int col   = lane & 15;           // C-layout column (N index)
    const int rbase = (lane >> 4) << 3;    // C-layout row base (0 or 8); == A k-offset
    const int hbase = hh << 8;

    // ---- cooperative staging (all source rows are contiguous) ----
    {
        const _Float16* tsrc = tnh + (b * Nn + n0) * En;
        for (int i = tid; i < 4 * 256; i += 256) sT[i] = tsrc[i];
        const _Float16* dsrc = dnh + (b * Mn + m0) * En;
        for (int i = tid; i < 16 * 256; i += 256) sD[i] = dsrc[i];
        const float* asrc = Ap + (b * Nn + n0) * Hn;
        for (int i = tid; i < 4 * 512; i += 256) sAp[i] = asrc[i];
        for (int i = tid; i < 512; i += 256) {
            sG[i] = gamma[i]; sBt[i] = beta[i]; sW2[i] = W2[i];
        }
    }

    v8f acc[16];
    const v8f vzero = {0.f, 0.f, 0.f, 0.f, 0.f, 0.f, 0.f, 0.f};
#pragma unroll
    for (int c = 0; c < 16; ++c) acc[c] = vzero;

    // ---- K loop: 8 chunks of 32 ----
    for (int kc = 0; kc < 8; ++kc) {
        __syncthreads();
        {   // stage 32KB of prepacked B for this K chunk (straight copy)
            const uint4* src = (const uint4*)(Bp + kc * 16384);
            uint4*       dst = (uint4*)sB;
#pragma unroll
            for (int r = 0; r < 8; ++r) dst[r * 256 + tid] = src[r * 256 + tid];
        }
        __syncthreads();

        // Build A operand in registers per the 16-bit 16x32 A layout:
        // lane half0: k = {0..7, 16..23}; half1: k = {8..15, 24..31} (+kc*32)
        v16h a;
        {
            const _Float16* tp = sT + nl  * En + kc * 32 + rbase;
            const _Float16* dp = sD + col * En + kc * 32 + rbase;
#pragma unroll
            for (int i = 0; i < 8; ++i) {
                float x = (float)tp[i]      - (float)dp[i];
                float y = (float)tp[16 + i] - (float)dp[16 + i];
                a[i]     = (_Float16)__builtin_fabsf(x);
                a[8 + i] = (_Float16)__builtin_fabsf(y);
            }
        }

#pragma unroll
        for (int c = 0; c < 16; ++c) {
            v16h bv = *(const v16h*)(sB + (((hh << 4) + c) << 9) + (lane << 4));
            acc[c] = __builtin_amdgcn_wmma_f32_16x16x32_f16(
                false, a, false, bv, (short)0, acc[c], false, false);
        }
    }

    // ---- epilogue pass 1: add precomputed parts, gather LN stats ----
    float s[8], s2[8];
#pragma unroll
    for (int j = 0; j < 8; ++j) { s[j] = 0.f; s2[j] = 0.f; }

    const float* DpB = Dp + (b * Mn + m0 + rbase) * Hn + hbase + col;
#pragma unroll
    for (int c = 0; c < 16; ++c) {
        float ap = sAp[nl * Hn + hbase + c * 16 + col];
#pragma unroll
        for (int j = 0; j < 8; ++j) {
            float v = acc[c][j] + ap + DpB[j * Hn + c * 16];
            acc[c][j] = v;
            s[j]  += v;
            s2[j] += v * v;
        }
    }
    // reduce across the 16 lanes of each C-layout half
#pragma unroll
    for (int j = 0; j < 8; ++j) {
#pragma unroll
        for (int m = 1; m < 16; m <<= 1) {
            s[j]  += __shfl_xor(s[j],  m, 32);
            s2[j] += __shfl_xor(s2[j], m, 32);
        }
    }
#pragma unroll
    for (int j = 0; j < 8; ++j)
        if ((lane & 15) == j) {
            sRed[wave][rbase + j][0] = s[j];
            sRed[wave][rbase + j][1] = s2[j];
        }
    __syncthreads();

    float mu[8], rs[8];
#pragma unroll
    for (int j = 0; j < 8; ++j) {
        int row = rbase + j;
        float T  = sRed[wave][row][0] + sRed[wave ^ 4][row][0];
        float T2 = sRed[wave][row][1] + sRed[wave ^ 4][row][1];
        float m_  = T * (1.0f / 512.0f);
        float var = T2 * (1.0f / 512.0f) - m_ * m_;
        mu[j] = m_;
        rs[j] = rsqrtf(var + 1e-5f);
    }

    // ---- epilogue pass 2: LN affine + SiLU + W2 dot ----
    float o[8];
#pragma unroll
    for (int j = 0; j < 8; ++j) o[j] = 0.f;
#pragma unroll
    for (int c = 0; c < 16; ++c) {
        int h = hbase + c * 16 + col;
        float g = sG[h], bt = sBt[h], w2 = sW2[h];
#pragma unroll
        for (int j = 0; j < 8; ++j) {
            float v = (acc[c][j] - mu[j]) * rs[j] * g + bt;
            v = v / (1.0f + __expf(-v));      // SiLU
            o[j] += v * w2;
        }
    }
#pragma unroll
    for (int j = 0; j < 8; ++j) {
#pragma unroll
        for (int m = 1; m < 16; m <<= 1) o[j] += __shfl_xor(o[j], m, 32);
    }
#pragma unroll
    for (int j = 0; j < 8; ++j)
        if ((lane & 15) == j) sOut[wave][rbase + j] = o[j];
    __syncthreads();

    if (wave < 4 && lane < 16) {
        int n = n0 + wave;
        float val = sOut[wave][lane] + sOut[wave + 4][lane] + b2[0];
        out[(b * Nn + n) * Mn + m0 + lane] = val;
    }
}

// ---------------------------------------------------------------------------
extern "C" void kernel_launch(void* const* d_in, const int* in_sizes, int n_in,
                              void* d_out, int out_size, void* d_ws, size_t ws_size,
                              hipStream_t stream) {
    const float* track = (const float*)d_in[0];
    const float* det   = (const float*)d_in[1];
    const float* W1    = (const float*)d_in[2];
    const float* b1    = (const float*)d_in[3];
    const float* gamma = (const float*)d_in[4];
    const float* beta  = (const float*)d_in[5];
    const float* W2    = (const float*)d_in[6];
    const float* b2    = (const float*)d_in[7];
    float* out = (float*)d_out;

    char* ws = (char*)d_ws;
    float*    tn  = (float*)(ws);                              // 1.00 MB
    float*    dn  = (float*)(ws + (1u << 20));                 // 1.00 MB
    float*    Ap  = (float*)(ws + (2u << 20));                 // 2.00 MB
    float*    Dp  = (float*)(ws + (4u << 20));                 // 2.00 MB
    _Float16* tnh = (_Float16*)(ws + (6u << 20));              // 0.50 MB
    _Float16* dnh = (_Float16*)(ws + (6u << 20) + (1u << 19)); // 0.50 MB
    _Float16* Bp  = (_Float16*)(ws + (7u << 20));              // 0.25 MB

    prep_norm<<<Bn * Nn, 256, 0, stream>>>(track, tn, tnh);
    prep_norm<<<Bn * Mn, 256, 0, stream>>>(det, dn, dnh);
    prep_part<<<Bn * Nn * 2, 256, 0, stream>>>(tn, W1, b1, Ap, 0,  1);
    prep_part<<<Bn * Mn * 2, 256, 0, stream>>>(dn, W1, b1, Dp, En, 0);
    prep_bpack<<<512, 256, 0, stream>>>(W1, Bp);

    tdsp_main<<<Bn * (Nn / 4) * (Mn / 16), 256, 0, stream>>>(
        tnh, dnh, Ap, Dp, Bp, gamma, beta, W2, b2, out);
}
